// ScoringFunction_3384434229924
// MI455X (gfx1250) — matrix operand order repaired and checked by reading, not compiled, not run on hardware
//
#include <hip/hip_runtime.h>

typedef _Float16 v16h __attribute__((ext_vector_type(16)));
typedef float    v8f  __attribute__((ext_vector_type(8)));
typedef float    v4f  __attribute__((ext_vector_type(4)));   // native vector for NT loads

// Problem constants (match reference setup_inputs)
constexpr int QTOK = 32;    // query tokens per query
constexpr int DIM  = 128;   // embedding dim (K)
constexpr int DLEN = 256;   // doc tokens per doc
constexpr int KSTEP = 32;   // K per WMMA
constexpr int NWAVE = 8;    // waves per block

__device__ __forceinline__ void cvt8(const v4f a, const v4f b, v16h& f, int off) {
    f[off + 0] = (_Float16)a[0]; f[off + 1] = (_Float16)a[1];
    f[off + 2] = (_Float16)a[2]; f[off + 3] = (_Float16)a[3];
    f[off + 4] = (_Float16)b[0]; f[off + 5] = (_Float16)b[1];
    f[off + 6] = (_Float16)b[2]; f[off + 7] = (_Float16)b[3];
}

// A fragment (16x32 f16): this lane needs 16 consecutive K values p[0..15].
// Query data is small and reused by 128 blocks -> regular temporal loads.
__device__ __forceinline__ v16h load_a_frag(const float* __restrict__ p) {
    const v4f a0 = *(const v4f*)(p + 0);
    const v4f a1 = *(const v4f*)(p + 4);
    const v4f a2 = *(const v4f*)(p + 8);
    const v4f a3 = *(const v4f*)(p + 12);
    v16h f; cvt8(a0, a1, f, 0); cvt8(a2, a3, f, 8);
    return f;
}

// B fragment (32x16 f16): this lane needs 16 consecutive K values p[0..15].
// Doc embeddings are 256MB of single-use streaming data vs 192MB L2 ->
// non-temporal loads so they don't evict the hot query tiles.
__device__ __forceinline__ v16h load_b_frag(const float* __restrict__ p) {
    const v4f a0 = __builtin_nontemporal_load((const v4f*)(p + 0));
    const v4f a1 = __builtin_nontemporal_load((const v4f*)(p + 4));
    const v4f a2 = __builtin_nontemporal_load((const v4f*)(p + 8));
    const v4f a3 = __builtin_nontemporal_load((const v4f*)(p + 12));
    v16h f; cvt8(a0, a1, f, 0); cvt8(a2, a3, f, 8);
    return f;
}

__global__ __launch_bounds__(256, 2)
void colbert_maxsim_kernel(const float* __restrict__ qemb,
                           const unsigned char* __restrict__ qmask,
                           const float* __restrict__ demb,
                           const unsigned char* __restrict__ dmask,
                           const int* __restrict__ num_docs_p,
                           float* __restrict__ out) {
    const int n    = blockIdx.x;            // doc index
    const int lane = threadIdx.x & 31;
    const int wave = threadIdx.x >> 5;
    const int col  = lane & 15;             // N / M position inside 16-wide tile
    const int hi   = lane >> 4;             // lane half (K-split selector)

    const int nd = *num_docs_p;             // uniform scalar
    const int b  = n / nd;                  // owning query

    const float* __restrict__ qbase = qemb + (size_t)b * QTOK * DIM;
    const float* __restrict__ dbase = demb + (size_t)n * DLEN * DIM;

    // This wave owns doc-token tiles nt0, nt1 (16 tokens each) x both M tiles.
    const int nt0 = wave * 2;
    const int nt1 = wave * 2 + 1;

    v8f c[2][2] = {};   // [mtile][ntile] f32 accumulators

#pragma unroll
    for (int ks = 0; ks < DIM / KSTEP; ++ks) {
        const int akoff = ks * KSTEP + hi * 16;   // A: lanes16-31 hold K+16..31
        const int bkoff = ks * KSTEP + hi * 16;   // B: lanes16-31 hold K+16..31

        // Query fragments for M-tiles 0,1 (rows mt*16 + col)
        const v16h a0 = load_a_frag(qbase + (size_t)(0 * 16 + col) * DIM + akoff);
        const v16h a1 = load_a_frag(qbase + (size_t)(1 * 16 + col) * DIM + akoff);
        // Doc fragments for the wave's two token tiles (row = token, contiguous K)
        const v16h b0 = load_b_frag(dbase + (size_t)(nt0 * 16 + col) * DIM + bkoff);
        const v16h b1 = load_b_frag(dbase + (size_t)(nt1 * 16 + col) * DIM + bkoff);

        c[0][0] = __builtin_amdgcn_wmma_f32_16x16x32_f16(false, a0, false, b0, (short)0, c[0][0], false, false);
        c[0][1] = __builtin_amdgcn_wmma_f32_16x16x32_f16(false, a0, false, b1, (short)0, c[0][1], false, false);
        c[1][0] = __builtin_amdgcn_wmma_f32_16x16x32_f16(false, a1, false, b0, (short)0, c[1][0], false, false);
        c[1][1] = __builtin_amdgcn_wmma_f32_16x16x32_f16(false, a1, false, b1, (short)0, c[1][1], false, false);
    }

    // ---- MaxSim reduction ----
    const float NEG = -3.402823466e38f;
    const bool v0 = __builtin_nontemporal_load(&dmask[(size_t)n * DLEN + nt0 * 16 + col]) != 0;
    const bool v1 = __builtin_nontemporal_load(&dmask[(size_t)n * DLEN + nt1 * 16 + col]) != 0;

    __shared__ float red[NWAVE * QTOK];

    // C/D layout: element r, lanes 0-15 -> M = r, lanes 16-31 -> M = 8 + r;
    // lane column = doc token (nt*16 + col). Mask invalid tokens, max the two
    // token tiles, then max-reduce across the 16 lanes of each half.
    #pragma unroll
    for (int mt = 0; mt < 2; ++mt) {
        float m[8];
        #pragma unroll
        for (int r = 0; r < 8; ++r) {
            const float x0 = v0 ? c[mt][0][r] : NEG;
            const float x1 = v1 ? c[mt][1][r] : NEG;
            m[r] = fmaxf(x0, x1);
        }
        #pragma unroll
        for (int xm = 1; xm <= 8; xm <<= 1) {
            #pragma unroll
            for (int r = 0; r < 8; ++r)
                m[r] = fmaxf(m[r], __shfl_xor(m[r], xm, 32));
        }
        if (col == 0) {
            const int qrow = mt * 16 + hi * 8;  // queries covered by this half
            #pragma unroll
            for (int r = 0; r < 8; ++r)
                red[wave * QTOK + qrow + r] = m[r];
        }
    }
    __syncthreads();

    // ---- Masked mean over query tokens (wave 0) ----
    if (wave == 0) {
        const int qi = lane;                     // query token 0..31
        float mx = red[qi];
        #pragma unroll
        for (int w = 1; w < NWAVE; ++w)
            mx = fmaxf(mx, red[w * QTOK + qi]);

        const bool qv = qmask[(size_t)b * QTOK + qi] != 0;
        float s   = qv ? mx  : 0.0f;
        float cnt = qv ? 1.0f : 0.0f;
        #pragma unroll
        for (int xm = 1; xm <= 16; xm <<= 1) {
            s   += __shfl_xor(s,   xm, 32);
            cnt += __shfl_xor(cnt, xm, 32);
        }
        if (lane == 0)
            out[n] = (cnt == 0.0f) ? 0.0f : s / cnt;
    }
}

extern "C" void kernel_launch(void* const* d_in, const int* in_sizes, int n_in,
                              void* d_out, int out_size, void* d_ws, size_t ws_size,
                              hipStream_t stream) {
    const float*         qemb  = (const float*)d_in[0];
    const unsigned char* qmask = (const unsigned char*)d_in[1];
    const float*         demb  = (const float*)d_in[2];
    const unsigned char* dmask = (const unsigned char*)d_in[3];
    const int*           ndocs = (const int*)d_in[4];
    float*               out   = (float*)d_out;

    const int N = out_size;  // number of docs (2048)
    colbert_maxsim_kernel<<<N, 256, 0, stream>>>(qemb, qmask, demb, dmask, ndocs, out);
}
